// FilterbankMaskPolicy_45105746542888
// MI455X (gfx1250) — matrix operand order repaired and checked by reading, not compiled, not run on hardware
//
#include <hip/hip_runtime.h>
#include <hip/hip_bf16.h>
#include <cstdint>
#include <cstddef>

// CDNA5 wave32 WMMA types
typedef __attribute__((ext_vector_type(16))) _Float16 v16h;
typedef __attribute__((ext_vector_type(8)))  float    v8f;

#define NBANKS   4096
#define PADW     2048
#define TILE_B   16
#define NTHREADS 256
#define NWAVES   8
#define NWORDS   (NBANKS / 32)   // 128 packed words per row

union AFrag { uint32_t u[8]; v16h h; };
union BFrag { uint4    q[2]; v16h h; };

__device__ __forceinline__ int reflect_idx(int t) {
    return (t < 0) ? -t : ((t < NBANKS) ? t : (2 * NBANKS - 2 - t));
}

__global__ __launch_bounds__(NTHREADS)
void fbmask_kernel(const float* __restrict__ logits,
                   const float* __restrict__ u,
                   const int*   __restrict__ shift,
                   float* __restrict__ masks,
                   float* __restrict__ log_probs)
{
    __shared__ uint32_t gbits[TILE_B][NWORDS];                         // 8 KB packed Bernoulli grid
    __shared__ float    plds[NBANKS];                                  // 16 KB sigmoid(logits)
    __shared__ _Float16 wlds[NBANKS] __attribute__((aligned(16)));     // 8 KB logits as f16 (WMMA B)
    __shared__ float    partial[NWAVES][TILE_B];
    __shared__ float    red[NTHREADS];
    __shared__ int      shifts[TILE_B];
    __shared__ float    csum;                                          // sum_k log_sigmoid(-logits[k])

    const int tid  = threadIdx.x;
    const int lane = tid & 31;
    const int wave = tid >> 5;
    const int half = lane >> 4;              // 0: lanes 0-15, 1: lanes 16-31
    const int r0   = blockIdx.x * TILE_B;

    if (tid < TILE_B) shifts[tid] = shift[r0 + tid];

    // ---- Phase 0: per-bank constants (p, f16 logits, logsig(-x) sum) ----
    float lacc = 0.0f;
    for (int k = tid; k < NBANKS; k += NTHREADS) {
        float x = logits[k];
        plds[k] = 1.0f / (1.0f + __expf(-x));
        wlds[k] = (_Float16)x;
        // log_sigmoid(-x) = -softplus(x), numerically stable
        lacc -= fmaxf(x, 0.0f) + log1pf(__expf(-fabsf(x)));
    }
    red[tid] = lacc;
    __syncthreads();
    if (tid == 0) {
        float s = 0.0f;
        for (int i = 0; i < NTHREADS; ++i) s += red[i];
        csum = s;
    }

    // ---- Phase 1: sample grid = (u < p), pack bits via wave32 ballot ----
    // Each wave-iteration: 4 batched coalesced b32 loads (4 words of one row),
    // 4 ballots, one ds_store_b128 of the packed uint4.
    for (int g = wave; g < (TILE_B * NWORDS) / 4; g += NWAVES) {   // 512 groups
        int row = g >> 5;                    // 32 groups of 4 words per row
        int w0  = (g & 31) * 4;
        int col = w0 * 32 + lane;
        const float* up = &u[(size_t)(r0 + row) * NBANKS + col];
        float u0 = up[0];
        float u1 = up[32];
        float u2 = up[64];
        float u3 = up[96];
        uint32_t m0 = __builtin_amdgcn_ballot_w32(u0 < plds[col]);
        uint32_t m1 = __builtin_amdgcn_ballot_w32(u1 < plds[col + 32]);
        uint32_t m2 = __builtin_amdgcn_ballot_w32(u2 < plds[col + 64]);
        uint32_t m3 = __builtin_amdgcn_ballot_w32(u3 < plds[col + 96]);
        if (lane == 0) {
            *(uint4*)&gbits[row][w0] = make_uint4(m0, m1, m2, m3);
        }
    }
    __syncthreads();

    // ---- Phase 2: log_probs dot via WMMA: D = grid(16x32) x logits_chunk(32x16) ----
    // A fragment (16-bit 16x32): lanes 0-15 row=lane, K {0..7,16..23};
    //                            lanes 16-31 row=lane-16, K {8..15,24..31}.
    // B fragment (16-bit 32x16): lanes 0-15 K {0..7,16..23}; lanes 16-31 K {8..15,24..31};
    //   logits replicated into every column -> all 16 D columns equal the row dot.
    v8f acc = {0.f, 0.f, 0.f, 0.f, 0.f, 0.f, 0.f, 0.f};
    for (int chunk = wave; chunk < NWORDS; chunk += NWAVES) {
        uint32_t word = gbits[lane & 15][chunk];
        uint32_t lo8  = (word >> (half * 8)) & 0xFFu;        // K groups 0-7 / 8-15
        uint32_t hi8  = (word >> (16 + half * 8)) & 0xFFu;   // K groups 16-23 / 24-31
        AFrag a;
#pragma unroll
        for (int i = 0; i < 4; ++i) {
            a.u[i]     = (((lo8 >> (2 * i)) & 1u) ? 0x00003C00u : 0u) |
                         (((lo8 >> (2 * i + 1)) & 1u) ? 0x3C000000u : 0u);
            a.u[4 + i] = (((hi8 >> (2 * i)) & 1u) ? 0x00003C00u : 0u) |
                         (((hi8 >> (2 * i + 1)) & 1u) ? 0x3C000000u : 0u);
        }
        BFrag b;
        const uint4* wp = (const uint4*)&wlds[chunk * 32 + half * 8];
        b.q[0] = wp[0];            // ds_load_b128 (LDS broadcast within half-wave)
        b.q[1] = wp[2];
        acc = __builtin_amdgcn_wmma_f32_16x16x32_f16(
                  /*neg_a=*/false, a.h, /*neg_b=*/false, b.h,
                  /*c_mod=*/(short)0, acc, /*reuse_a=*/false, /*reuse_b=*/false);
    }
    // Column 0 of D: lane 0 holds rows 0-7 in acc[0..7], lane 16 holds rows 8-15.
    if ((lane & 15) == 0) {
#pragma unroll
        for (int r = 0; r < 8; ++r) partial[wave][half * 8 + r] = acc[r];
    }
    __syncthreads();

    if (tid < TILE_B) {
        float s = csum;
#pragma unroll
        for (int w = 0; w < NWAVES; ++w) s += partial[w][tid];
        log_probs[r0 + tid] = s;
    }

    // ---- Phase 3: masks via reflect-padded shifted gather, float4 stores ----
    // Each lane produces 4 consecutive output floats -> global_store_b128,
    // wave writes 512 contiguous bytes per store instruction.
    for (int it = 0; it < (TILE_B * NBANKS) / (NTHREADS * 4); ++it) {  // 64 iters
        int e   = it * NTHREADS + tid;
        int row = e >> 10;                  // 1024 float4 per row
        int j0  = (e & 1023) << 2;
        int s   = shifts[row];
        float4 v;
        {
            int m0 = reflect_idx(s + j0 + 0 - PADW);
            int m1 = reflect_idx(s + j0 + 1 - PADW);
            int m2 = reflect_idx(s + j0 + 2 - PADW);
            int m3 = reflect_idx(s + j0 + 3 - PADW);
            v.x = (float)((gbits[row][m0 >> 5] >> (m0 & 31)) & 1u);
            v.y = (float)((gbits[row][m1 >> 5] >> (m1 & 31)) & 1u);
            v.z = (float)((gbits[row][m2 >> 5] >> (m2 & 31)) & 1u);
            v.w = (float)((gbits[row][m3 >> 5] >> (m3 & 31)) & 1u);
        }
        float4* mrow = (float4*)&masks[(size_t)(r0 + row) * NBANKS];
        mrow[e & 1023] = v;
    }
}

extern "C" void kernel_launch(void* const* d_in, const int* in_sizes, int n_in,
                              void* d_out, int out_size, void* d_ws, size_t ws_size,
                              hipStream_t stream) {
    (void)n_in; (void)out_size; (void)d_ws; (void)ws_size;
    const float* logits = (const float*)d_in[0];   // (NB,)  f32
    const float* u      = (const float*)d_in[1];   // (B,NB) f32
    const int*   shift  = (const int*)  d_in[2];   // (B,)   i32
    const int    NB = in_sizes[0];                 // 4096
    const int    B  = in_sizes[2];                 // 8192

    float* out       = (float*)d_out;
    float* masks     = out;                        // B*1*NB floats
    float* log_probs = out + (size_t)B * NB;       // B floats

    dim3 grid(B / TILE_B);
    fbmask_kernel<<<grid, NTHREADS, 0, stream>>>(logits, u, shift, masks, log_probs);
}
